// MixtureOfExperts_37237366456694
// MI455X (gfx1250) — compile-verified, matrix operand order checked by reading, not run on hardware
//
#include <hip/hip_runtime.h>
#include <hip/hip_bf16.h>

// ---------------------------------------------------------------------------
// MoE + Cayley-projection pipeline for gfx1250 (MI455X).
// All heavy GEMMs run on v_wmma_f32_16x16x32_bf16 (bf16 in, f32 accumulate),
// with a 2-deep LDS pipeline so next-tile global loads overlap the WMMAs.
// ---------------------------------------------------------------------------

typedef __bf16 bf16_t;
typedef __attribute__((ext_vector_type(2)))  __bf16 v2bf;
typedef __attribute__((ext_vector_type(16))) __bf16 v16bf;
typedef __attribute__((ext_vector_type(8)))  float  v8f;

#define B_SZ  4096
#define D_IN  1024
#define H1_   4096
#define H2_   4096
#define D_OUT 1024
#define K_EXP 8
#define G_H   512
#define SUB   128   // D_OUT / K_EXP

__device__ __forceinline__ bf16_t f2bf(float f) { return (bf16_t)f; }
__device__ __forceinline__ float  bf2f(bf16_t b) { return (float)b; }

// Pack two floats into one dword of two bf16 (lo, hi) — single v_cvt_pk_bf16_f32.
__device__ __forceinline__ unsigned pk2bf(float lo, float hi) {
  union { v2bf v; unsigned u; } o;
#if __has_builtin(__builtin_amdgcn_cvt_pk_bf16_f32)
  o.v = __builtin_amdgcn_cvt_pk_bf16_f32(lo, hi);
#else
  o.v.x = (bf16_t)lo; o.v.y = (bf16_t)hi;
#endif
  return o.u;
}

// Load one 16x32 bf16 WMMA operand fragment from LDS (row-major [..][32]).
// Lane l: row = l&15 (caller bakes row in), K chunks at (l>>4)*8 and +16.
__device__ __forceinline__ v16bf load_frag(const bf16_t* p) {
  union { uint4 u[2]; v16bf v; } r;
  r.u[0] = *(const uint4*)p;
  r.u[1] = *(const uint4*)(p + 16);
  return r.v;
}

// ---------------------------------------------------------------------------
// Tiled WMMA GEMM:  C[z] = act( A[z] (MxK bf16, row-major) @ B[z] (KxN f32,
// row-major, converted to bf16 on the fly) + bias[z] ), C stored bf16.
// Block tile 128x128x32, 256 threads = 8 wave32; each wave: 2x4 16x16 tiles.
// Double-buffered LDS (32 KB): fetch k-tile i+1 while computing on tile i.
// All of M, N, K must be multiples of 128/128/32 (true for every call here).
// ---------------------------------------------------------------------------
__global__ __launch_bounds__(256) void gemm_wmma_bf16(
    const bf16_t* __restrict__ A, const float* __restrict__ Bm,
    const float* __restrict__ bias, bf16_t* __restrict__ C,
    int M, int N, int Kd, int lda, int ldb, int ldc,
    long sA, long sB, long sBias, long sC, int relu)
{
  __shared__ __attribute__((aligned(16))) bf16_t As[2][128][32];
  __shared__ __attribute__((aligned(16))) bf16_t Bs[2][128][32];  // B^T tile [n][k]

  const int tid  = threadIdx.x;
  const int lane = tid & 31;
  const int wid  = tid >> 5;
  const int wm   = wid & 3;    // 4 wave-rows x 32
  const int wn   = wid >> 2;   // 2 wave-cols x 64
  const int m0   = blockIdx.y * 128;
  const int n0   = blockIdx.x * 128;
  const int z    = blockIdx.z;

  const bf16_t* Ae = A  + (long)z * sA;
  const float*  Be = Bm + (long)z * sB;
  const float*  be = bias ? (bias + (long)z * sBias) : nullptr;
  bf16_t*       Ce = C  + (long)z * sC;

  const int arow = tid >> 1, aseg = tid & 1;  // A tile: 128 rows x 2 segs of 16
  const int bkp  = tid >> 4;                  // 0..15 -> k-pair (2*bkp, 2*bkp+1)
  const int bcg  = tid & 15;                  // 0..15 -> cols bcg*8 .. +7
  const int l15  = lane & 15;
  const int kb   = (lane >> 4) * 8;

  v8f acc[2][4] = {};

  // Per-thread staging registers (next k-tile in flight during compute)
  uint4 a0, a1;
  float fb0[8], fb1[8];

  auto fetch = [&](int kt) {
    const bf16_t* ag = Ae + (long)(m0 + arow) * lda + kt + aseg * 16;
    a0 = *(const uint4*)ag;
    a1 = *(const uint4*)(ag + 8);
    const float* b0 = Be + (long)(kt + 2 * bkp) * ldb + n0 + bcg * 8;
    const float* b1 = b0 + ldb;
    *(float4*)&fb0[0] = *(const float4*)(b0);
    *(float4*)&fb0[4] = *(const float4*)(b0 + 4);
    *(float4*)&fb1[0] = *(const float4*)(b1);
    *(float4*)&fb1[4] = *(const float4*)(b1 + 4);
    if (kt + 32 < Kd) {                     // CDNA5 global_prefetch_b8
      __builtin_prefetch(ag + 32, 0, 1);
      __builtin_prefetch(b0 + (long)32 * ldb, 0, 1);
    }
  };
  auto stage = [&](int buf) {
    *(uint4*)&As[buf][arow][aseg * 16]     = a0;
    *(uint4*)&As[buf][arow][aseg * 16 + 8] = a1;
#pragma unroll
    for (int j = 0; j < 8; ++j)             // one v_cvt_pk + one ds_store_b32 each
      *(unsigned*)&Bs[buf][bcg * 8 + j][2 * bkp] = pk2bf(fb0[j], fb1[j]);
  };

  fetch(0);
  stage(0);
  __syncthreads();

  const int nt = Kd >> 5;
  for (int i = 0; i < nt; ++i) {
    if (i + 1 < nt) fetch((i + 1) << 5);    // overlap with WMMAs below

    const int cur = i & 1;
    v16bf af[2], bfr[4];
#pragma unroll
    for (int sm = 0; sm < 2; ++sm)
      af[sm] = load_frag(&As[cur][wm * 32 + sm * 16 + l15][kb]);
#pragma unroll
    for (int sn = 0; sn < 4; ++sn)
      bfr[sn] = load_frag(&Bs[cur][wn * 64 + sn * 16 + l15][kb]);

#pragma unroll
    for (int sm = 0; sm < 2; ++sm)
#pragma unroll
      for (int sn = 0; sn < 4; ++sn)
        acc[sm][sn] = __builtin_amdgcn_wmma_f32_16x16x32_bf16(
            false, af[sm], false, bfr[sn], (short)0, acc[sm][sn], false, false);

    if (i + 1 < nt) {                       // uniform condition -> legal barrier
      stage((i + 1) & 1);
      __syncthreads();
    }
  }

  // --- epilogue: bias + ReLU + bf16 store (C/D layout: VGPR j -> row j / j+8) ---
#pragma unroll
  for (int sm = 0; sm < 2; ++sm) {
    const int rbase = m0 + wm * 32 + sm * 16 + ((lane < 16) ? 0 : 8);
#pragma unroll
    for (int sn = 0; sn < 4; ++sn) {
      const int col = n0 + wn * 64 + sn * 16 + l15;
      const float bv = be ? be[col] : 0.0f;
#pragma unroll
      for (int j = 0; j < 8; ++j) {
        float v = acc[sm][sn][j] + bv;
        if (relu) v = v > 0.0f ? v : 0.0f;
        Ce[(long)(rbase + j) * ldc + col] = f2bf(v);
      }
    }
  }
}

// ---------------------------------------------------------------------------
__global__ void cvt_f32_to_bf16(const float* __restrict__ in,
                                bf16_t* __restrict__ out, long n)
{
  long i = (long)blockIdx.x * blockDim.x + threadIdx.x;
  if (i < n) out[i] = f2bf(in[i]);
}

// Gating layer 2 + softmax: g[b,k] = softmax_k(relu_h @ Wg2 + bg2)
__global__ void gating2_softmax(const bf16_t* __restrict__ hg,
                                const float* __restrict__ Wg2,
                                const float* __restrict__ bg2,
                                float* __restrict__ g, int Bsz)
{
  int b = blockIdx.x * blockDim.x + threadIdx.x;
  if (b >= Bsz) return;
  float acc[K_EXP];
#pragma unroll
  for (int k = 0; k < K_EXP; ++k) acc[k] = bg2[k];
  const bf16_t* hrow = hg + (long)b * G_H;
  for (int d = 0; d < G_H; ++d) {
    float h = bf2f(hrow[d]);
#pragma unroll
    for (int k = 0; k < K_EXP; ++k) acc[k] += h * Wg2[d * K_EXP + k];
  }
  float m = acc[0];
#pragma unroll
  for (int k = 1; k < K_EXP; ++k) m = fmaxf(m, acc[k]);
  float s = 0.0f;
#pragma unroll
  for (int k = 0; k < K_EXP; ++k) { acc[k] = __expf(acc[k] - m); s += acc[k]; }
  float inv = 1.0f / s;
#pragma unroll
  for (int k = 0; k < K_EXP; ++k) g[(long)b * K_EXP + k] = acc[k] * inv;
}

// Build augmented system [ I-S | I+S ],  S = A - A^T  (1024 x 2048)
__global__ void build_aug(const float* __restrict__ A, float* __restrict__ Mg)
{
  int j = blockIdx.x * 256 + threadIdx.x;    // 0..2047
  int i = blockIdx.y;                         // 0..1023
  if (j < D_OUT) {
    float s = A[(long)i * D_OUT + j] - A[(long)j * D_OUT + i];
    Mg[(long)i * 2048 + j] = ((i == j) ? 1.0f : 0.0f) - s;
  } else {
    int jj = j - D_OUT;
    float s = A[(long)i * D_OUT + jj] - A[(long)jj * D_OUT + i];
    Mg[(long)i * 2048 + j] = ((i == jj) ? 1.0f : 0.0f) + s;
  }
}

// Gauss-Jordan (no pivoting; I-S is positive-real so diagonals stay ~1).
// One workgroup, 1024 threads, barrier per pivot.  Q = (I-S)^{-1}(I+S).
__global__ __launch_bounds__(1024) void gauss_jordan(float* __restrict__ Mg,
                                                     float* __restrict__ Q)
{
  const int r = threadIdx.x;
  float* rowr = Mg + (long)r * 2048;
  for (int p = 0; p < D_OUT; ++p) {
    __syncthreads();
    const float* rowp = Mg + (long)p * 2048;
    if (r != p) {
      float f = rowr[p] / rowp[p];
      float4*       dst = (float4*)rowr;
      const float4* src = (const float4*)rowp;
      for (int j = p >> 2; j < 512; ++j) {
        float4 a = dst[j], b = src[j];
        a.x -= f * b.x; a.y -= f * b.y; a.z -= f * b.z; a.w -= f * b.w;
        dst[j] = a;
      }
    }
  }
  __syncthreads();
  float inv = 1.0f / rowr[r];
  for (int j = 0; j < D_OUT; ++j)
    Q[(long)r * D_OUT + j] = rowr[D_OUT + j] * inv;
}

// 1024x1024 transpose (Qt = Q^T), LDS-tiled.
__global__ __launch_bounds__(256) void transpose1024(const float* __restrict__ Q,
                                                     float* __restrict__ Qt)
{
  __shared__ float t[32][33];
  int bx = blockIdx.x * 32, by = blockIdx.y * 32;
  int x = threadIdx.x & 31, y = threadIdx.x >> 5;   // 32 x 8
#pragma unroll
  for (int i = 0; i < 32; i += 8)
    t[y + i][x] = Q[(long)(by + y + i) * 1024 + bx + x];
  __syncthreads();
#pragma unroll
  for (int i = 0; i < 32; i += 8)
    Qt[(long)(bx + y + i) * 1024 + by + x] = t[x][y + i];
}

// Fused gate-weighted combine + output head: out[b] = (sum_k g[b,k] V[k,b,:]) . Wo + bo
__global__ __launch_bounds__(256) void moe_head(const bf16_t* __restrict__ V,
                                                const float* __restrict__ g,
                                                const float* __restrict__ Wo,
                                                const float* __restrict__ bo,
                                                float* __restrict__ out, int Bsz)
{
  __shared__ float red[256];
  int b = blockIdx.x;
  float gk[K_EXP];
#pragma unroll
  for (int k = 0; k < K_EXP; ++k) gk[k] = g[(long)b * K_EXP + k];
  float acc = 0.0f;
  for (int d = threadIdx.x; d < D_OUT; d += 256) {
    float v = 0.0f;
#pragma unroll
    for (int k = 0; k < K_EXP; ++k)
      v += gk[k] * bf2f(V[((long)k * Bsz + b) * D_OUT + d]);
    acc += v * Wo[d];
  }
  red[threadIdx.x] = acc;
  __syncthreads();
  for (int s = 128; s > 0; s >>= 1) {
    if (threadIdx.x < s) red[threadIdx.x] += red[threadIdx.x + s];
    __syncthreads();
  }
  if (threadIdx.x == 0) out[b] = red[0] + bo[0];
}

// ---------------------------------------------------------------------------
extern "C" void kernel_launch(void* const* d_in, const int* in_sizes, int n_in,
                              void* d_out, int out_size, void* d_ws, size_t ws_size,
                              hipStream_t stream)
{
  const float* x   = (const float*)d_in[0];
  const float* W1  = (const float*)d_in[1];
  const float* b1  = (const float*)d_in[2];
  const float* W2  = (const float*)d_in[3];
  const float* b2  = (const float*)d_in[4];
  const float* W3  = (const float*)d_in[5];
  const float* b3  = (const float*)d_in[6];
  const float* Wg1 = (const float*)d_in[7];
  const float* bg1 = (const float*)d_in[8];
  const float* Wg2 = (const float*)d_in[9];
  const float* bg2 = (const float*)d_in[10];
  const float* Ap  = (const float*)d_in[11];
  const float* Wo  = (const float*)d_in[12];
  const float* bo  = (const float*)d_in[13];
  float* out = (float*)d_out;

  // ---- workspace layout (stream-ordered aliasing; peak ~590 MB) ----
  char* ws = (char*)d_ws;
  const size_t SZ_XB = (size_t)B_SZ * D_IN * 2;              //   8 MiB
  const size_t SZ_HG = (size_t)B_SZ * G_H * 2;               //   4 MiB
  const size_t SZ_G  = (size_t)B_SZ * K_EXP * 4;             // 128 KiB
  const size_t SZ_H  = (size_t)K_EXP * B_SZ * H1_ * 2;       // 256 MiB
  const size_t OFF_XB = 0;
  const size_t OFF_HG = OFF_XB + SZ_XB;
  const size_t OFF_G  = OFF_HG + SZ_HG;
  const size_t OFF_H1 = OFF_G + SZ_G;                        // h1; later V aliases here
  const size_t OFF_H2 = OFF_H1 + SZ_H;                       // h2; later Cayley scratch
  const size_t OFF_UB = OFF_H2 + SZ_H;                       // U (bf16, K x B x D_OUT)
  // Aliased into h2 region after layer-3 completes:
  const size_t OFF_AUG = OFF_H2;                             //   8 MiB (1024 x 2048 f32)
  const size_t OFF_Q   = OFF_AUG + (size_t)D_OUT * 2048 * 4; //   4 MiB
  const size_t OFF_QT  = OFF_Q   + (size_t)D_OUT * D_OUT * 4;//   4 MiB
  const size_t OFF_CO  = OFF_QT  + (size_t)D_OUT * D_OUT * 4;//   8 MiB (K x B x 128 bf16)
  const size_t OFF_V   = OFF_H1;                             //  64 MiB (aliases dead h1)

  bf16_t* xb     = (bf16_t*)(ws + OFF_XB);
  bf16_t* hg     = (bf16_t*)(ws + OFF_HG);
  float*  g      = (float*) (ws + OFF_G);
  bf16_t* h1     = (bf16_t*)(ws + OFF_H1);
  bf16_t* h2     = (bf16_t*)(ws + OFF_H2);
  bf16_t* Ub     = (bf16_t*)(ws + OFF_UB);
  float*  aug    = (float*) (ws + OFF_AUG);
  float*  Q      = (float*) (ws + OFF_Q);
  float*  Qt     = (float*) (ws + OFF_QT);
  bf16_t* coords = (bf16_t*)(ws + OFF_CO);
  bf16_t* Vb     = (bf16_t*)(ws + OFF_V);

  // 1) x -> bf16
  cvt_f32_to_bf16<<<(B_SZ * D_IN) / 256, 256, 0, stream>>>(x, xb, (long)B_SZ * D_IN);

  // 2) gating layer 1: hg = relu(x @ Wg1 + bg1)   (4096 x 512)
  gemm_wmma_bf16<<<dim3(G_H / 128, B_SZ / 128, 1), 256, 0, stream>>>(
      xb, Wg1, bg1, hg, B_SZ, G_H, D_IN, D_IN, G_H, G_H, 0, 0, 0, 0, 1);

  // 3) gating layer 2 + softmax
  gating2_softmax<<<B_SZ / 256, 256, 0, stream>>>(hg, Wg2, bg2, g, B_SZ);

  // 4) expert layer 1: h1[k] = relu(x @ W1[k] + b1[k])
  gemm_wmma_bf16<<<dim3(H1_ / 128, B_SZ / 128, K_EXP), 256, 0, stream>>>(
      xb, W1, b1, h1, B_SZ, H1_, D_IN, D_IN, H1_, H1_,
      0L, (long)D_IN * H1_, (long)H1_, (long)B_SZ * H1_, 1);

  // 5) expert layer 2: h2[k] = relu(h1[k] @ W2[k] + b2[k])
  gemm_wmma_bf16<<<dim3(H2_ / 128, B_SZ / 128, K_EXP), 256, 0, stream>>>(
      h1, W2, b2, h2, B_SZ, H2_, H1_, H1_, H2_, H2_,
      (long)B_SZ * H1_, (long)H1_ * H2_, (long)H2_, (long)B_SZ * H2_, 1);

  // 6) expert layer 3: U[k] = h2[k] @ W3[k] + b3[k]
  gemm_wmma_bf16<<<dim3(D_OUT / 128, B_SZ / 128, K_EXP), 256, 0, stream>>>(
      h2, W3, b3, Ub, B_SZ, D_OUT, H2_, H2_, D_OUT, D_OUT,
      (long)B_SZ * H2_, (long)H2_ * D_OUT, (long)D_OUT, (long)B_SZ * D_OUT, 0);

  // 7) Cayley: Q = (I-S)^{-1}(I+S)   (h2 region is dead now)
  build_aug<<<dim3(2048 / 256, D_OUT), 256, 0, stream>>>(Ap, aug);
  gauss_jordan<<<1, 1024, 0, stream>>>(aug, Q);
  transpose1024<<<dim3(32, 32), 256, 0, stream>>>(Q, Qt);

  // 8) coords[k] = U[k] @ Bq[k]; Bq[k] = Q[:, k*128:(k+1)*128] => strided view
  gemm_wmma_bf16<<<dim3(SUB / 128, B_SZ / 128, K_EXP), 256, 0, stream>>>(
      Ub, Q, nullptr, coords, B_SZ, SUB, D_OUT, D_OUT, D_OUT, SUB,
      (long)B_SZ * D_OUT, (long)SUB, 0L, (long)B_SZ * SUB, 0);

  // 9) V[k] = coords[k] @ Bq[k]^T; Bq[k]^T = Qt[k*128:(k+1)*128, :]
  gemm_wmma_bf16<<<dim3(D_OUT / 128, B_SZ / 128, K_EXP), 256, 0, stream>>>(
      coords, Qt, nullptr, Vb, B_SZ, D_OUT, SUB, SUB, D_OUT, D_OUT,
      (long)B_SZ * SUB, (long)SUB * D_OUT, 0L, (long)B_SZ * D_OUT, 0);

  // 10) out[b] = (sum_k g[b,k] V[k,b,:]) . Wo + bo
  moe_head<<<B_SZ, 256, 0, stream>>>(Vb, g, Wo, bo, out, B_SZ);
}
